// nGCNN_1735166787761
// MI455X (gfx1250) — matrix operand order repaired and checked by reading, not compiled, not run on hardware
//
#include <hip/hip_runtime.h>
#include <stdint.h>

#define NND 20000
#define NED 320000

typedef __attribute__((ext_vector_type(16))) __bf16 v16bf;
typedef __attribute__((ext_vector_type(8)))  float  v8f;

__device__ __forceinline__ uint32_t f2bf_bits(float f) {
  uint32_t u = __builtin_bit_cast(uint32_t, f);
  return (u + 0x7FFFu + ((u >> 16) & 1u)) >> 16;   // RNE to bf16, low 16 bits
}
__device__ __forceinline__ uint32_t pack2bf(float lo, float hi) {
  return f2bf_bits(lo) | (f2bf_bits(hi) << 16);
}
__device__ __forceinline__ __bf16 f2bf(float f) {
  uint16_t s = (uint16_t)f2bf_bits(f);
  __bf16 b;
  __builtin_memcpy(&b, &s, 2);
  return b;
}

__global__ void k_zero(float* __restrict__ p, int n) {
  int i = blockIdx.x * blockDim.x + threadIdx.x;
  if (i < n) p[i] = 0.f;
}

// agg[dst[e]][4q:4q+4) += ew[e] * h[src[e]][4q:4q+4) ; one thread per (edge, quad)
__global__ void k_scatter4(const int* __restrict__ src, const int* __restrict__ dst,
                           const float* __restrict__ ew, const float* __restrict__ h,
                           float* __restrict__ agg, int E, int dinLog2) {
  int idx = blockIdx.x * blockDim.x + threadIdx.x;
  int qLog = dinLog2 - 2;
  int e = idx >> qLog;
  if (e >= E) return;
  int q = idx & ((1 << qLog) - 1);
  int s = src[e], d = dst[e];
  float w = ew[e];
  const float4 v = *(const float4*)(h + ((size_t)s << dinLog2) + q * 4);
  float* ap = agg + ((size_t)d << dinLog2) + q * 4;
  __hip_atomic_fetch_add(ap + 0, w * v.x, __ATOMIC_RELAXED, __HIP_MEMORY_SCOPE_AGENT);
  __hip_atomic_fetch_add(ap + 1, w * v.y, __ATOMIC_RELAXED, __HIP_MEMORY_SCOPE_AGENT);
  __hip_atomic_fetch_add(ap + 2, w * v.z, __ATOMIC_RELAXED, __HIP_MEMORY_SCOPE_AGENT);
  __hip_atomic_fetch_add(ap + 3, w * v.w, __ATOMIC_RELAXED, __HIP_MEMORY_SCOPE_AGENT);
}

// Acat[i][0:din) = bf16(agg[i]) ; Acat[i][din:2din) = bf16(h[i]) ; 4 elems/thread
__global__ void k_pack_a4(const float* __restrict__ agg, const float* __restrict__ h,
                          __bf16* __restrict__ Acat, int N, int dinLog2) {
  int idx = blockIdx.x * blockDim.x + threadIdx.x;        // quad index
  int qLog = dinLog2 - 1;                                 // quads per row = K2/4 = din/2
  if (idx >= (N << qLog)) return;
  int i = idx >> qLog;
  int kq = idx & ((1 << qLog) - 1);
  int k = kq * 4;
  int din = 1 << dinLog2;
  const float* sp = (k < din) ? (agg + ((size_t)i << dinLog2) + k)
                              : (h + ((size_t)i << dinLog2) + (k - din));
  float4 v = *(const float4*)sp;
  uint2 out;
  out.x = pack2bf(v.x, v.y);
  out.y = pack2bf(v.z, v.w);
  *(uint2*)(Acat + ((size_t)idx << 2)) = out;
}

// Wcat[n][0:din) = bf16(Wrel[n]) ; Wcat[n][din:2din) = bf16(Wroot[n])
__global__ void k_pack_w(const float* __restrict__ Wrel, const float* __restrict__ Wroot,
                         __bf16* __restrict__ Wcat, int dout, int dinLog2) {
  int idx = blockIdx.x * blockDim.x + threadIdx.x;
  int K2Log = dinLog2 + 1;
  int din = 1 << dinLog2;
  if (idx >= (dout << K2Log)) return;
  int n = idx >> K2Log;
  int k = idx & ((1 << K2Log) - 1);
  float v = (k < din) ? Wrel[((size_t)n << dinLog2) + k]
                      : Wroot[((size_t)n << dinLog2) + (k - din)];
  Wcat[idx] = f2bf(v);
}

// out[m][n] = sum_k Acat[m][k] * Wcat[n][k] + bias[n], optional ReLU.
// block = 128 threads = 4 waves; wave tile = 16 rows x (16*NT) cols.
// A fragment is loaded once per k-step and reused across NT WMMAs.
template <int NT>
__global__ __launch_bounds__(128) void k_gemm_wmma(
    const __bf16* __restrict__ Acat, const __bf16* __restrict__ Wcat,
    const float* __restrict__ bias, float* __restrict__ out,
    int K2, int dinLog2, int dout, int relu) {
  // padded LDS stripe: 16 rows, (K2 + 8) halves stride to break bank conflicts
  __shared__ __bf16 sA[16 * (1024 + 8)];
  const int din = K2 >> 1;              // words per unpadded row
  const int spw = din + 4;              // padded stride in 32-bit words
  const int m0 = blockIdx.x * 16;
  const int tid = threadIdx.x;

  // stage 16 x K2 bf16 stripe of A into LDS (word copies)
  {
    const uint32_t* g = (const uint32_t*)(Acat + (size_t)m0 * K2);
    uint32_t* sdst = (uint32_t*)sA;
    const int total = 16 << dinLog2;    // 16 rows * din words
    for (int i = tid; i < total; i += 128) {
      int row = i >> dinLog2;
      int w = i & (din - 1);
      sdst[row * spw + w] = g[((size_t)row << dinLog2) + w];
    }
  }
  __syncthreads();

  const int wave = tid >> 5;
  const int lane = tid & 31;
  const int n0 = (blockIdx.y * 4 + wave) * (16 * NT);
  const int col = lane & 15;
  const int hi = lane >> 4;             // 0 for lanes 0-15, 1 for 16-31
  const int lb = hi << 3;               // C: row offset 0 / 8
  const int koff = hi << 4;             // B: K base offset 0 / 16 (halves)

  v8f c[NT];
#pragma unroll
  for (int t = 0; t < NT; ++t) c[t] = (v8f){0.f, 0.f, 0.f, 0.f, 0.f, 0.f, 0.f, 0.f};

  const uint32_t* srow = (const uint32_t*)sA + (size_t)col * spw;
  const __bf16* bbase[NT];
#pragma unroll
  for (int t = 0; t < NT; ++t)
    bbase[t] = Wcat + (size_t)(n0 + t * 16 + col) * K2 + koff;

  for (int kb = 0; kb < K2; kb += 32) {
    v16bf a;
    uint32_t* ap = (uint32_t*)&a;
#pragma unroll
    for (int v = 0; v < 8; ++v) {
      // A layout (matches B): lanes 0-15 hold K {0..15}, lanes 16-31 hold
      // K {16..31}; pairs of halves are contiguous -> one b32 LDS read.
      int kh = kb + 2 * v + koff;   // even half index
      ap[v] = srow[kh >> 1];
    }
#pragma unroll
    for (int t = 0; t < NT; ++t) {
      // B layout (matches A): lane holds col n, 16 contiguous K at koff
      v16bf b = *(const v16bf*)(bbase[t] + kb);
      c[t] = __builtin_amdgcn_wmma_f32_16x16x32_bf16(false, a, false, b,
                                                     (short)0, c[t], false, false);
    }
  }

#pragma unroll
  for (int t = 0; t < NT; ++t) {
    const int n = n0 + t * 16 + col;
    const float bn = bias[n];
#pragma unroll
    for (int r = 0; r < 8; ++r) {
      int m = m0 + r + lb;               // C layout: VGPR r -> row r (+8 hi half)
      float v = c[t][r] + bn;
      if (relu) v = fmaxf(v, 0.f);
      out[(size_t)m * dout + n] = v;
    }
  }
}

__global__ void k_bn_stats(const float* __restrict__ h, float* __restrict__ mean,
                           float* __restrict__ rstd, int N, int C) {
  __shared__ float ss[256];
  __shared__ float ss2[256];
  int cidx = blockIdx.x;
  float s = 0.f, s2 = 0.f;
  for (int i = threadIdx.x; i < N; i += blockDim.x) {
    float v = h[(size_t)i * C + cidx];
    s += v; s2 += v * v;
  }
  ss[threadIdx.x] = s; ss2[threadIdx.x] = s2;
  __syncthreads();
  for (int o = 128; o > 0; o >>= 1) {
    if (threadIdx.x < o) {
      ss[threadIdx.x] += ss[threadIdx.x + o];
      ss2[threadIdx.x] += ss2[threadIdx.x + o];
    }
    __syncthreads();
  }
  if (threadIdx.x == 0) {
    float mu = ss[0] / (float)N;
    float var = ss2[0] / (float)N - mu * mu;
    mean[cidx] = mu;
    rstd[cidx] = rsqrtf(var + 1e-5f);
  }
}

__global__ void k_bn_apply(float* __restrict__ h, const float* __restrict__ mean,
                           const float* __restrict__ rstd, const float* __restrict__ gamma,
                           const float* __restrict__ beta, int N, int C) {
  int idx = blockIdx.x * blockDim.x + threadIdx.x;
  if (idx >= N * C) return;
  int cidx = idx % C;
  float v = (h[idx] - mean[cidx]) * rstd[cidx] * gamma[cidx] + beta[cidx];
  h[idx] = fmaxf(v, 0.f);
}

extern "C" void kernel_launch(void* const* d_in, const int* in_sizes, int n_in,
                              void* d_out, int out_size, void* d_ws, size_t ws_size,
                              hipStream_t stream) {
  (void)in_sizes; (void)n_in; (void)out_size; (void)ws_size;
  const int N = NND, E = NED;

  // setup_inputs() dict order:
  const float* x     = (const float*)d_in[0];
  const int*   ei    = (const int*)d_in[1];     // [2][E]
  const float* ew    = (const float*)d_in[2];
  const float* gamma = (const float*)d_in[3];
  const float* beta  = (const float*)d_in[4];
  const float* W1r = (const float*)d_in[5];  const float* b1 = (const float*)d_in[6];  const float* W1o = (const float*)d_in[7];
  const float* W2r = (const float*)d_in[8];  const float* b2 = (const float*)d_in[9];  const float* W2o = (const float*)d_in[10];
  const float* W3r = (const float*)d_in[11]; const float* b3 = (const float*)d_in[12]; const float* W3o = (const float*)d_in[13];
  const float* W4r = (const float*)d_in[14]; const float* b4 = (const float*)d_in[15]; const float* W4o = (const float*)d_in[16];

  const int* src = ei;
  const int* dst = ei + E;

  // workspace layout (all offsets multiples of 32B)
  char* ws = (char*)d_ws;
  const size_t SZ = (size_t)N * 512 * sizeof(float);   // 40.96 MB
  float*  bufA   = (float*)(ws + 0 * SZ);
  float*  bufB   = (float*)(ws + 1 * SZ);
  float*  agg    = (float*)(ws + 2 * SZ);
  __bf16* Acat   = (__bf16*)(ws + 3 * SZ);             // [N][2*din] bf16, <= 40.96 MB
  __bf16* Wcat   = (__bf16*)(ws + 4 * SZ);             // <= 1 MB, reused per layer
  float*  bnMean = (float*)(ws + 4 * SZ + (1 << 20));
  float*  bnRstd = bnMean + 512;

  auto layer = [&](const float* hin, int dinLog2,
                   const float* Wrel, const float* Wroot, const float* bias,
                   int dout, int relu, float* hout) {
    const int din = 1 << dinLog2;
    const int K2 = din * 2;
    const int nAgg = N * din;
    k_zero<<<(nAgg + 255) / 256, 256, 0, stream>>>(agg, nAgg);
    const int nS = E * (din / 4);
    k_scatter4<<<(nS + 255) / 256, 256, 0, stream>>>(src, dst, ew, hin, agg, E, dinLog2);
    const int nP = N * (K2 / 4);
    k_pack_a4<<<(nP + 255) / 256, 256, 0, stream>>>(agg, hin, Acat, N, dinLog2);
    const int nW = dout * K2;
    k_pack_w<<<(nW + 255) / 256, 256, 0, stream>>>(Wrel, Wroot, Wcat, dout, dinLog2);
    if (dout >= 256) {
      dim3 grid(N / 16, dout / 256);     // 4 waves x (16x64) per block
      k_gemm_wmma<4><<<grid, 128, 0, stream>>>(Acat, Wcat, bias, hout, K2, dinLog2, dout, relu);
    } else {
      dim3 grid(N / 16, dout / 64);      // 4 waves x (16x16) per block
      k_gemm_wmma<1><<<grid, 128, 0, stream>>>(Acat, Wcat, bias, hout, K2, dinLog2, dout, relu);
    }
  };

  layer(x,    7, W1r, W1o, b1, 256, 1, bufA);   // 128 -> 256, ReLU
  layer(bufA, 8, W2r, W2o, b2, 512, 1, bufB);   // 256 -> 512, ReLU
  layer(bufB, 9, W3r, W3o, b3, 256, 0, bufA);   // 512 -> 256
  k_bn_stats<<<256, 256, 0, stream>>>(bufA, bnMean, bnRstd, N, 256);
  k_bn_apply<<<(N * 256 + 255) / 256, 256, 0, stream>>>(bufA, bnMean, bnRstd, gamma, beta, N, 256);
  layer(bufA, 8, W4r, W4o, b4, 64, 0, (float*)d_out);  // 256 -> 64
}
